// MultiHeadAttention_80977313399158
// MI455X (gfx1250) — compile-verified, matrix operand order checked by reading, not run on hardware
//
#include <hip/hip_runtime.h>

typedef __attribute__((ext_vector_type(16))) __bf16 v16bf;
typedef __attribute__((ext_vector_type(8)))  float  v8f;

#define BATCH  2
#define SEQ    2048
#define CMODEL 768
#define NHEAD  12
#define DHEAD  64
#define MROWS  (BATCH * SEQ)     /* 4096 */
#define N3     (3 * CMODEL)      /* 2304 */

union FragU {
  v16bf v;
  uint4 q[2];
};

__device__ __forceinline__ unsigned short f32_to_bf16(float f) {
  union { float f; unsigned int u; } x;
  x.f = f;
  unsigned int u = x.u;
  u += 0x7FFFu + ((u >> 16) & 1u);   // round-to-nearest-even
  return (unsigned short)(u >> 16);
}

// Load a 16x32 (A) or 32x16 (B, as [N][K] row-major) bf16 WMMA fragment from
// global.  Lane l holds row/col (l & 15), K chunks {half*8..+7, 16+half*8..+7}.
__device__ __forceinline__ v16bf load_frag_g(const unsigned short* __restrict__ p,
                                             int row, int stride, int kbase, int lane) {
  const unsigned short* rp = p + (size_t)row * stride + kbase + ((lane >> 4) << 3);
  FragU f;
  f.q[0] = *(const uint4*)(rp);
  f.q[1] = *(const uint4*)(rp + 16);
  return f.v;
}

// Same fragment gather but from LDS (ds_load_b128 path).
__device__ __forceinline__ v16bf load_frag_lds(const unsigned short* p,
                                               int row, int stride, int kbase, int lane) {
  const unsigned short* rp = p + row * stride + kbase + ((lane >> 4) << 3);
  FragU f;
  __builtin_memcpy(&f.q[0], rp, 16);
  __builtin_memcpy(&f.q[1], rp + 16, 16);
  return f.v;
}

__device__ __forceinline__ v8f wmma_bf16(v16bf a, v16bf b, v8f c) {
  return __builtin_amdgcn_wmma_f32_16x16x32_bf16(false, a, false, b, (short)0, c,
                                                 false, false);
}

// ---- CDNA5 async global->LDS copy (ASYNCcnt-tracked), ISA 08_async_tensor §4
__device__ __forceinline__ void async_load_b128(unsigned lds_off, const void* gaddr) {
  asm volatile("global_load_async_to_lds_b128 %0, %1, off"
               :: "v"(lds_off), "v"(gaddr) : "memory");
}
__device__ __forceinline__ void wait_async0() {
  asm volatile("s_wait_asynccnt 0x0" ::: "memory");
}

__device__ __forceinline__ float redmax16(float v) {
  v = fmaxf(v, __shfl_xor(v, 1, 32));
  v = fmaxf(v, __shfl_xor(v, 2, 32));
  v = fmaxf(v, __shfl_xor(v, 4, 32));
  v = fmaxf(v, __shfl_xor(v, 8, 32));
  return v;
}
__device__ __forceinline__ float redsum16(float v) {
  v += __shfl_xor(v, 1, 32);
  v += __shfl_xor(v, 2, 32);
  v += __shfl_xor(v, 4, 32);
  v += __shfl_xor(v, 8, 32);
  return v;
}

// ---------------- Kernel 1: f32 -> bf16 cast ----------------
__global__ void cast_f32_to_bf16_k(const float* __restrict__ s,
                                   unsigned short* __restrict__ d, int n) {
  int i = blockIdx.x * blockDim.x + threadIdx.x;
  if (i < n) d[i] = f32_to_bf16(s[i]);
}

// ---------------- Kernel 2: fused QKV GEMM + RoPE ----------------
// One wave computes a 32(M) x 64(N) tile; the 64 N-columns are exactly one
// head's D of one of {Q,K,V}.  Q pre-scaled by 1/sqrt(D)=0.125.
// Q,K stored [BH][T][64] bf16; V stored transposed [BH][64][T] bf16.
__global__ void __launch_bounds__(128)
qkv_rope_k(const unsigned short* __restrict__ xbf,
           const unsigned short* __restrict__ wbf,
           unsigned short* __restrict__ Qb,
           unsigned short* __restrict__ Kb,
           unsigned short* __restrict__ Vt) {
  const int lane = threadIdx.x & 31;
  const int l16  = lane & 15;
  const int half = lane >> 4;
  const int wid  = blockIdx.x * 4 + (threadIdx.x >> 5);
  const int mtile = wid / 36;    // 0..127, 32 rows each
  const int ntile = wid % 36;

  const v8f Z = {0.f, 0.f, 0.f, 0.f, 0.f, 0.f, 0.f, 0.f};
  v8f acc[8];
#pragma unroll
  for (int i = 0; i < 8; ++i) acc[i] = Z;

  const int arow0 = mtile * 32 + l16;
  const int arow1 = arow0 + 16;
  const int nbase = ntile * 64;
  const unsigned short* xrow = xbf + (size_t)arow0 * CMODEL;

  for (int kk = 0; kk < CMODEL; kk += 32) {
    __builtin_prefetch((const void*)(xrow + kk + 32), 0, 1);
    v16bf a0 = load_frag_g(xbf, arow0, CMODEL, kk, lane);
    v16bf a1 = load_frag_g(xbf, arow1, CMODEL, kk, lane);
#pragma unroll
    for (int jn = 0; jn < 4; ++jn) {
      v16bf b = load_frag_g(wbf, nbase + jn * 16 + l16, CMODEL, kk, lane);
      acc[jn]     = wmma_bf16(a0, b, acc[jn]);
      acc[4 + jn] = wmma_bf16(a1, b, acc[4 + jn]);
    }
  }

  const int sel = ntile / 12;   // 0=Q 1=K 2=V
  const int h   = ntile % 12;

  if (sel < 2) {
    // RoPE: pair (d, d+32) -> (acc j, acc j+2) within each row group
    const float lg  = 0.4152410118609203f;         // log2(10000)/32
    const float if0 = exp2f(-(float)l16 * lg);
    const float if1 = exp2f(-(float)(l16 + 16) * lg);
    const float qscale = (sel == 0) ? 0.125f : 1.0f;
    unsigned short* dst = (sel == 0) ? Qb : Kb;
#pragma unroll
    for (int g = 0; g < 2; ++g) {
#pragma unroll
      for (int r = 0; r < 8; ++r) {
        const int mg = mtile * 32 + g * 16 + r + half * 8;
        const int bb = mg >> 11;
        const int t  = mg & 2047;
        float sn0, cs0, sn1, cs1;
        __sincosf((float)t * if0, &sn0, &cs0);
        __sincosf((float)t * if1, &sn1, &cs1);
        float lo0 = acc[g * 4 + 0][r], lo1 = acc[g * 4 + 1][r];
        float hi0 = acc[g * 4 + 2][r], hi1 = acc[g * 4 + 3][r];
        float o0 = (lo0 * cs0 - hi0 * sn0) * qscale;
        float o1 = (lo1 * cs1 - hi1 * sn1) * qscale;
        float o2 = (hi0 * cs0 + lo0 * sn0) * qscale;
        float o3 = (hi1 * cs1 + lo1 * sn1) * qscale;
        size_t base = ((size_t)(bb * 12 + h) * SEQ + t) * DHEAD;
        dst[base +      l16] = f32_to_bf16(o0);
        dst[base + 16 + l16] = f32_to_bf16(o1);
        dst[base + 32 + l16] = f32_to_bf16(o2);
        dst[base + 48 + l16] = f32_to_bf16(o3);
      }
    }
  } else {
#pragma unroll
    for (int g = 0; g < 2; ++g) {
#pragma unroll
      for (int r = 0; r < 8; ++r) {
        const int mg = mtile * 32 + g * 16 + r + half * 8;
        const int bb = mg >> 11;
        const int t  = mg & 2047;
        size_t hb = (size_t)(bb * 12 + h) * DHEAD;
#pragma unroll
        for (int jn = 0; jn < 4; ++jn)
          Vt[(hb + jn * 16 + l16) * SEQ + t] = f32_to_bf16(acc[g * 4 + jn][r]);
      }
    }
  }
}

// ---------------- Kernel 3: causal flash attention ----------------
// Block = 4 waves = 4 consecutive 16-row q-tiles of one (b,h).  K/V 32-key
// tiles are staged into LDS once per block with async global->LDS copies
// (double-buffered on ASYNCcnt), then consumed by all 4 waves.
__device__ __forceinline__ void issue_kv_tile(const unsigned short* Kp,
                                              const unsigned short* Vp,
                                              int kb, unsigned klds, unsigned vlds,
                                              int tid) {
#pragma unroll
  for (int i = 0; i < 2; ++i) {
    const int c = tid + i * 128;                       // 0..255 chunks of 16B
    // K tile: 32 rows x 64 d  (row-major, 8 chunks/row)
    const unsigned short* kg = Kp + (size_t)(kb + (c >> 3)) * DHEAD + ((c & 7) << 3);
    async_load_b128(klds + (unsigned)c * 16u, (const void*)kg);
    // V tile: 64 d-rows x 32 keys (row-major, 4 chunks/row)
    const unsigned short* vg = Vp + (size_t)(c >> 2) * SEQ + kb + ((c & 3) << 3);
    async_load_b128(vlds + (unsigned)c * 16u, (const void*)vg);
  }
}

__global__ void __launch_bounds__(128)
flash_attn_k(const unsigned short* __restrict__ Qb,
             const unsigned short* __restrict__ Kb,
             const unsigned short* __restrict__ Vt,
             unsigned short* __restrict__ Ob) {
  __shared__ __align__(16) unsigned short Kt[2][32 * 64];
  __shared__ __align__(16) unsigned short Vw[2][64 * 32];
  __shared__ __align__(16) unsigned short Pl[4][16 * 32];

  const int tid  = threadIdx.x;
  const int lane = tid & 31;
  const int l16  = lane & 15;
  const int half = lane >> 4;
  const int warp = tid >> 5;

  const int bh = blockIdx.x >> 5;    // 0..23
  const int qg = blockIdx.x & 31;    // 0..31 (groups of 4 q-tiles)
  const int qi = qg * 4 + warp;      // 0..127
  const int qb = qi * 16;

  const unsigned short* Qp = Qb + (size_t)bh * SEQ * DHEAD;
  const unsigned short* Kp = Kb + (size_t)bh * SEQ * DHEAD;
  const unsigned short* Vp = Vt + (size_t)bh * DHEAD * SEQ;

  const v16bf qa0 = load_frag_g(Qp, qb + l16, DHEAD, 0,  lane);
  const v16bf qa1 = load_frag_g(Qp, qb + l16, DHEAD, 32, lane);

  const v8f Z = {0.f, 0.f, 0.f, 0.f, 0.f, 0.f, 0.f, 0.f};
  v8f o[4];
  o[0] = Z; o[1] = Z; o[2] = Z; o[3] = Z;
  float mr[8], lr[8];
#pragma unroll
  for (int r = 0; r < 8; ++r) { mr[r] = -1.0e30f; lr[r] = 0.f; }

  const int nkt     = (qi >> 1) + 1;   // this wave's causal tile count
  const int nkt_max = 2 * qg + 2;      // block-wide tile count
  unsigned short* pl = Pl[warp];

  const unsigned klds[2] = { (unsigned)(uintptr_t)&Kt[0][0],
                             (unsigned)(uintptr_t)&Kt[1][0] };
  const unsigned vlds[2] = { (unsigned)(uintptr_t)&Vw[0][0],
                             (unsigned)(uintptr_t)&Vw[1][0] };

  issue_kv_tile(Kp, Vp, 0, klds[0], vlds[0], tid);

  for (int kt = 0; kt < nkt_max; ++kt) {
    wait_async0();            // our async copies for buf[kt&1] have landed
    __syncthreads();          // ... and everyone else's too
    if (kt + 1 < nkt_max) {
      const int nb = (kt + 1) & 1;
      issue_kv_tile(Kp, Vp, (kt + 1) * 32, klds[nb], vlds[nb], tid);
    }
    if (kt < nkt) {           // wave-uniform: EXEC stays all-1s for WMMA
      const int kb = kt * 32;
      const unsigned short* Kl = Kt[kt & 1];
      const unsigned short* Vl = Vw[kt & 1];
      v8f s0 = Z, s1 = Z;
      {
        v16bf k0a = load_frag_lds(Kl, l16,      DHEAD, 0,  lane);
        v16bf k0b = load_frag_lds(Kl, l16,      DHEAD, 32, lane);
        v16bf k1a = load_frag_lds(Kl, 16 + l16, DHEAD, 0,  lane);
        v16bf k1b = load_frag_lds(Kl, 16 + l16, DHEAD, 32, lane);
        s0 = wmma_bf16(qa0, k0a, s0);
        s0 = wmma_bf16(qa1, k0b, s0);
        s1 = wmma_bf16(qa0, k1a, s1);
        s1 = wmma_bf16(qa1, k1b, s1);
      }
      if (kt == nkt - 1) {    // diagonal tile: causal mask
        const int kg0 = kb + l16, kg1 = kb + 16 + l16;
#pragma unroll
        for (int r = 0; r < 8; ++r) {
          const int qgl = qb + r + half * 8;
          if (kg0 > qgl) s0[r] = -3.0e38f;
          if (kg1 > qgl) s1[r] = -3.0e38f;
        }
      }
      float alpha[8];
#pragma unroll
      for (int r = 0; r < 8; ++r) {
        float rm   = redmax16(fmaxf(s0[r], s1[r]));
        float mnew = fmaxf(mr[r], rm);
        alpha[r]   = __expf(mr[r] - mnew);
        mr[r]      = mnew;
        float p0 = __expf(s0[r] - mnew);
        float p1 = __expf(s1[r] - mnew);
        s0[r] = p0; s1[r] = p1;
        lr[r] = lr[r] * alpha[r] + redsum16(p0 + p1);
      }
#pragma unroll
      for (int jn = 0; jn < 4; ++jn)
#pragma unroll
        for (int r = 0; r < 8; ++r) o[jn][r] *= alpha[r];

      // P (C-layout) -> LDS row-major 16x32 bf16 -> A-fragment layout
#pragma unroll
      for (int r = 0; r < 8; ++r) {
        const int rl = r + half * 8;
        pl[rl * 32 +      l16] = f32_to_bf16(s0[r]);
        pl[rl * 32 + 16 + l16] = f32_to_bf16(s1[r]);
      }
      FragU pf;
      {
        const unsigned short* pr = pl + l16 * 32 + half * 8;
        __builtin_memcpy(&pf.q[0], pr, 16);
        __builtin_memcpy(&pf.q[1], pr + 16, 16);
      }
#pragma unroll
      for (int jn = 0; jn < 4; ++jn) {
        v16bf vf = load_frag_lds(Vl, jn * 16 + l16, 32, 0, lane);
        o[jn] = wmma_bf16(pf.v, vf, o[jn]);
      }
    }
    __syncthreads();          // protect buf before next overwrite
  }

  const int b = bh / 12, h = bh % 12;
  float inv[8];
#pragma unroll
  for (int r = 0; r < 8; ++r) inv[r] = 1.0f / lr[r];
#pragma unroll
  for (int r = 0; r < 8; ++r) {
    const int t = qb + r + half * 8;
    size_t base = ((size_t)b * SEQ + t) * CMODEL + h * 64;
#pragma unroll
    for (int jn = 0; jn < 4; ++jn)
      Ob[base + jn * 16 + l16] = f32_to_bf16(o[jn][r] * inv[r]);
  }
}

// ---------------- Kernel 4: output projection (32x64 tile/wave) ----------------
__global__ void __launch_bounds__(128)
out_proj_k(const unsigned short* __restrict__ Ob,
           const unsigned short* __restrict__ wbf,
           float* __restrict__ out) {
  const int lane = threadIdx.x & 31;
  const int l16  = lane & 15;
  const int half = lane >> 4;
  const int wid  = blockIdx.x * 4 + (threadIdx.x >> 5);
  const int mtile = wid / 12;   // 0..127
  const int ntile = wid % 12;

  const v8f Z = {0.f, 0.f, 0.f, 0.f, 0.f, 0.f, 0.f, 0.f};
  v8f acc[8];
#pragma unroll
  for (int i = 0; i < 8; ++i) acc[i] = Z;

  const int arow0 = mtile * 32 + l16;
  const int arow1 = arow0 + 16;
  const int nbase = ntile * 64;
  const unsigned short* orow = Ob + (size_t)arow0 * CMODEL;

  for (int kk = 0; kk < CMODEL; kk += 32) {
    __builtin_prefetch((const void*)(orow + kk + 32), 0, 1);
    v16bf a0 = load_frag_g(Ob, arow0, CMODEL, kk, lane);
    v16bf a1 = load_frag_g(Ob, arow1, CMODEL, kk, lane);
#pragma unroll
    for (int jn = 0; jn < 4; ++jn) {
      v16bf b = load_frag_g(wbf, nbase + jn * 16 + l16, CMODEL, kk, lane);
      acc[jn]     = wmma_bf16(a0, b, acc[jn]);
      acc[4 + jn] = wmma_bf16(a1, b, acc[4 + jn]);
    }
  }
#pragma unroll
  for (int g = 0; g < 2; ++g) {
#pragma unroll
    for (int r = 0; r < 8; ++r) {
      const int mg = mtile * 32 + g * 16 + r + half * 8;
#pragma unroll
      for (int jn = 0; jn < 4; ++jn)
        out[(size_t)mg * CMODEL + nbase + jn * 16 + l16] = acc[g * 4 + jn][r];
    }
  }
}

extern "C" void kernel_launch(void* const* d_in, const int* in_sizes, int n_in,
                              void* d_out, int out_size, void* d_ws, size_t ws_size,
                              hipStream_t stream) {
  (void)in_sizes; (void)n_in; (void)out_size; (void)ws_size;
  const float* x     = (const float*)d_in[0];
  const float* w_qkv = (const float*)d_in[1];
  const float* w_out = (const float*)d_in[2];
  float* out = (float*)d_out;

  char* ws = (char*)d_ws;
  size_t off = 0;
  unsigned short* xbf    = (unsigned short*)(ws + off); off += (size_t)MROWS * CMODEL * 2;
  unsigned short* wqkvbf = (unsigned short*)(ws + off); off += (size_t)N3 * CMODEL * 2;
  unsigned short* woutbf = (unsigned short*)(ws + off); off += (size_t)CMODEL * CMODEL * 2;
  unsigned short* Qb     = (unsigned short*)(ws + off); off += (size_t)BATCH * NHEAD * SEQ * DHEAD * 2;
  unsigned short* Kb     = (unsigned short*)(ws + off); off += (size_t)BATCH * NHEAD * SEQ * DHEAD * 2;
  unsigned short* Vt     = (unsigned short*)(ws + off); off += (size_t)BATCH * NHEAD * SEQ * DHEAD * 2;
  unsigned short* Ob     = (unsigned short*)(ws + off); off += (size_t)MROWS * CMODEL * 2;

  int n;
  n = MROWS * CMODEL;
  cast_f32_to_bf16_k<<<(n + 255) / 256, 256, 0, stream>>>(x, xbf, n);
  n = N3 * CMODEL;
  cast_f32_to_bf16_k<<<(n + 255) / 256, 256, 0, stream>>>(w_qkv, wqkvbf, n);
  n = CMODEL * CMODEL;
  cast_f32_to_bf16_k<<<(n + 255) / 256, 256, 0, stream>>>(w_out, woutbf, n);

  // 128 M-tiles x 36 N-tiles = 4608 waves, 4 waves/block
  qkv_rope_k<<<1152, 128, 0, stream>>>(xbf, wqkvbf, Qb, Kb, Vt);
  // 24 (B*H) x 32 q-groups = 768 blocks (4 q-tiles each)
  flash_attn_k<<<768, 128, 0, stream>>>(Qb, Kb, Vt, Ob);
  // 128 M-tiles x 12 N-tiles = 1536 waves
  out_proj_k<<<384, 128, 0, stream>>>(Ob, woutbf, out);
}
